// SimpleRNN_62955630625426
// MI455X (gfx1250) — matrix operand order repaired and checked by reading, not compiled, not run on hardware
//
#include <hip/hip_runtime.h>
#include <hip/hip_bf16.h>

#define BATCH 32
#define SEQT  1024
#define HID   1024
#define LAYERS 4

typedef __attribute__((ext_vector_type(16))) __bf16 v16bf;
typedef __attribute__((ext_vector_type(8)))  float  v8f;

// ---------------------------------------------------------------- helpers

__device__ __forceinline__ unsigned short f2bf(float f) {
    unsigned u = __float_as_uint(f);
    unsigned r = u + 0x7FFFu + ((u >> 16) & 1u);   // round-to-nearest-even
    return (unsigned short)(r >> 16);
}

// A-operand tile load (16x32 bf16, ISA 7.12.2 layout): per lane two 16B groups.
// rowp points at this lane's row start (already includes row*stride).
__device__ __forceinline__ v16bf load_a(const unsigned short* rowp, int c, int alo) {
    union { uint4 q[2]; v16bf v; } u;
    u.q[0] = *(const uint4*)(rowp + c * 32 + alo);        // K = c*32 + alo + 0..7
    u.q[1] = *(const uint4*)(rowp + c * 32 + 16 + alo);   // K = c*32 + 16 + alo + 0..7
    return u.v;
}

// B-operand tile load from pre-packed buffer: 32B contiguous per lane.
__device__ __forceinline__ v16bf load_b(const unsigned short* p) {
    union { uint4 q[2]; v16bf v; } u;
    u.q[0] = *(const uint4*)(p);
    u.q[1] = *(const uint4*)(p + 8);
    return u.v;
}

__device__ __forceinline__ v8f wmma_bf16(v16bf a, v16bf b, v8f c) {
    return __builtin_amdgcn_wmma_f32_16x16x32_bf16(false, a, false, b, (short)0, c, false, false);
}

// ---------------------------------------------------------------- kernels

// fp32 -> bf16 elementwise (x -> seq buffer)
__global__ void cvt_f32_bf16_kernel(const float* __restrict__ in,
                                    unsigned short* __restrict__ out, int n) {
    for (int i = blockIdx.x * blockDim.x + threadIdx.x; i < n; i += gridDim.x * blockDim.x)
        out[i] = f2bf(in[i]);
}

// Repack row-major [layers][K=1024][N=1024] fp32 weights into the WMMA
// B-operand per-lane layout, bf16:
//   out[((l*64 + j)*32 + c)*512 + lane*16 + t]  <- W[l][c*32 + (t>>3)*16 + (lane>>4)*8 + (t&7)][j*16 + (lane&15)]
__global__ void repack_b_kernel(const float* __restrict__ W,
                                unsigned short* __restrict__ out, int layers) {
    const int per_layer = HID * HID;            // 64*32*512
    const int total = layers * per_layer;
    for (int e = blockIdx.x * blockDim.x + threadIdx.x; e < total; e += gridDim.x * blockDim.x) {
        int l  = e / per_layer;
        int r  = e % per_layer;
        int j  = r / 16384;            // ntile
        int r2 = r % 16384;
        int c  = r2 / 512;             // k-chunk
        int s  = r2 % 512;
        int lane = s / 16;
        int t    = s % 16;
        int k = c * 32 + (t >> 3) * 16 + (lane >> 4) * 8 + (t & 7);
        int n = j * 16 + (lane & 15);
        out[e] = f2bf(W[((size_t)l * HID + k) * HID + n]);
    }
}

// xp = seq @ Wi_l + b_l   ([32768,1024] @ [1024,1024] -> fp32)
// grid (2048 Mtiles, 2), block 256 (8 waves); wave handles 4 N-tiles.
// Software-pipelined: chunk c+1 loads are issued before chunk c's WMMAs.
__global__ void proj_kernel(const unsigned short* __restrict__ seq,
                            const unsigned short* __restrict__ Bp,
                            const float* __restrict__ bias,
                            float* __restrict__ xp) {
    const int lane = threadIdx.x & 31;
    const int wave = threadIdx.x >> 5;
    const int mtile = blockIdx.x;
    const int nb0 = blockIdx.y * 32 + wave * 4;
    const int alo = (lane >> 4) * 8;

    const unsigned short* arow = seq + (size_t)(mtile * 16 + (lane & 15)) * HID;
    const unsigned short* bbase = Bp + (size_t)nb0 * 32 * 512 + lane * 16;

    v8f acc[4];
    #pragma unroll
    for (int j = 0; j < 4; ++j) acc[j] = (v8f){0.f,0.f,0.f,0.f,0.f,0.f,0.f,0.f};

    v16bf a_cur = load_a(arow, 0, alo);
    v16bf b_cur[4];
    #pragma unroll
    for (int j = 0; j < 4; ++j) b_cur[j] = load_b(bbase + (size_t)j * 32 * 512);

    #pragma unroll
    for (int c = 0; c < 32; ++c) {
        const int cn = (c < 31) ? c + 1 : 31;           // compile-time per iteration
        v16bf a_nxt = load_a(arow, cn, alo);
        v16bf b_nxt[4];
        #pragma unroll
        for (int j = 0; j < 4; ++j)
            b_nxt[j] = load_b(bbase + ((size_t)j * 32 + cn) * 512);
        #pragma unroll
        for (int j = 0; j < 4; ++j)
            acc[j] = wmma_bf16(a_cur, b_cur[j], acc[j]);
        a_cur = a_nxt;
        #pragma unroll
        for (int j = 0; j < 4; ++j) b_cur[j] = b_nxt[j];
    }

    const int nlo = lane & 15;
    const int mhi = (lane >> 4) * 8;
    #pragma unroll
    for (int j = 0; j < 4; ++j) {
        int col = (nb0 + j) * 16 + nlo;
        float bv = bias[col];
        #pragma unroll
        for (int r = 0; r < 8; ++r) {
            int m = mtile * 16 + mhi + r;
            xp[(size_t)m * HID + col] = acc[j][r] + bv;
        }
    }
}

__global__ void init_barrier_kernel(unsigned* bar) {
    if (threadIdx.x < 4) bar[threadIdx.x] = 0u;
}

// Recurrent scan: h_t = tanh(xp_t + h_{t-1} @ Wh). 16 blocks x 256.
// Two independent 8-block groups (one per 16-batch-row M-tile); each wave owns
// one 16-col N-tile and keeps its Wh slice (256 VGPRs of bf16) in registers.
// xp loads for step t are issued before the WMMA chain; A is double-buffered.
__global__ void scan_kernel(const float* __restrict__ xp,
                            const unsigned short* __restrict__ Bp,
                            unsigned short* __restrict__ seq_out,
                            unsigned short* __restrict__ hprev,
                            unsigned* __restrict__ bar) {
    const int lane = threadIdx.x & 31;
    const int wave = threadIdx.x >> 5;
    const int g    = blockIdx.x >> 3;          // group / M-tile (0..1)
    const int nblk = blockIdx.x & 7;
    const int ntile = nblk * 8 + wave;         // 0..63
    const int alo = (lane >> 4) * 8;

    // persistent B registers: Wh[:, ntile*16 .. +16) in WMMA layout
    v16bf Breg[32];
    #pragma unroll
    for (int c = 0; c < 32; ++c)
        Breg[c] = load_b(Bp + ((size_t)ntile * 32 + c) * 512 + lane * 16);

    // zero this group's h rows (h0 = 0); every block zeroes exactly what it reads
    unsigned* hz = (unsigned*)(hprev + (size_t)g * 16 * HID);
    for (int i = threadIdx.x; i < 16 * HID / 2; i += 256) hz[i] = 0u;
    __threadfence();
    __syncthreads();

    const int nlo = lane & 15;
    const int mhi = (lane >> 4) * 8;
    const int col = ntile * 16 + nlo;
    const unsigned short* arow = hprev + (size_t)(g * 16 + (lane & 15)) * HID;
    unsigned* cnt   = bar + g;
    unsigned* phase = bar + 2 + g;

    for (int t = 0; t < SEQT; ++t) {
        // issue the strided xp loads first so they overlap the WMMA chain
        float xv[8];
        #pragma unroll
        for (int r = 0; r < 8; ++r) {
            int b = g * 16 + mhi + r;
            xv[r] = xp[((size_t)b * SEQT + t) * HID + col];
        }

        v8f acc0 = (v8f){0.f,0.f,0.f,0.f,0.f,0.f,0.f,0.f};
        v8f acc1 = (v8f){0.f,0.f,0.f,0.f,0.f,0.f,0.f,0.f};
        v16bf a_cur = load_a(arow, 0, alo);
        #pragma unroll
        for (int c = 0; c < 32; ++c) {
            const int cn = (c < 31) ? c + 1 : 31;
            v16bf a_nxt = load_a(arow, cn, alo);
            if (c & 1) acc1 = wmma_bf16(a_cur, Breg[c], acc1);
            else       acc0 = wmma_bf16(a_cur, Breg[c], acc0);
            a_cur = a_nxt;
        }
        v8f acc = acc0 + acc1;

        #pragma unroll
        for (int r = 0; r < 8; ++r) {
            int b = g * 16 + mhi + r;
            size_t idx = ((size_t)b * SEQT + t) * HID + col;
            float z = xv[r] + acc[r];
            float hv = tanhf(z);
            unsigned short hb = f2bf(hv);
            hprev[(size_t)b * HID + col] = hb;
            seq_out[idx] = hb;
            if (t + 1 < SEQT) __builtin_prefetch(&xp[idx + HID], 0, 1);  // global_prefetch_b8
        }

        __threadfence();           // make h stores device-visible before signaling
        __syncthreads();
        if (threadIdx.x == 0) {
            unsigned old = __hip_atomic_fetch_add(cnt, 1u, __ATOMIC_ACQ_REL,
                                                  __HIP_MEMORY_SCOPE_AGENT);
            if ((old & 7u) == 7u) {
                __hip_atomic_fetch_add(phase, 1u, __ATOMIC_RELEASE,
                                       __HIP_MEMORY_SCOPE_AGENT);
            } else {
                while (__hip_atomic_load(phase, __ATOMIC_ACQUIRE,
                                         __HIP_MEMORY_SCOPE_AGENT) < (unsigned)(t + 1))
                    __builtin_amdgcn_s_sleep(1);
            }
        }
        __syncthreads();
        __threadfence();           // acquire-side: don't reuse stale h lines
    }
}

// out = seq[:, T-1, :] @ fcW + fcb   ([32,1024]@[1024,1024] -> fp32)
__global__ void fc_kernel(const unsigned short* __restrict__ seq,
                          const unsigned short* __restrict__ Bp,
                          const float* __restrict__ fcb,
                          float* __restrict__ out) {
    const int lane = threadIdx.x & 31;
    const int wave = threadIdx.x >> 5;
    const int tile = blockIdx.x * 8 + wave;    // 0..127
    const int mtile = tile >> 6;
    const int ntile = tile & 63;
    const int alo = (lane >> 4) * 8;

    const unsigned short* arow =
        seq + ((size_t)(mtile * 16 + (lane & 15)) * SEQT + (SEQT - 1)) * HID;
    const unsigned short* bbase = Bp + (size_t)ntile * 32 * 512 + lane * 16;

    v8f acc = (v8f){0.f,0.f,0.f,0.f,0.f,0.f,0.f,0.f};
    v16bf a_cur = load_a(arow, 0, alo);
    v16bf b_cur = load_b(bbase);
    #pragma unroll
    for (int c = 0; c < 32; ++c) {
        const int cn = (c < 31) ? c + 1 : 31;
        v16bf a_nxt = load_a(arow, cn, alo);
        v16bf b_nxt = load_b(bbase + (size_t)cn * 512);
        acc = wmma_bf16(a_cur, b_cur, acc);
        a_cur = a_nxt;
        b_cur = b_nxt;
    }

    const int nlo = lane & 15;
    const int mhi = (lane >> 4) * 8;
    int col = ntile * 16 + nlo;
    float bv = fcb[col];
    #pragma unroll
    for (int r = 0; r < 8; ++r) {
        int m = mtile * 16 + mhi + r;
        out[(size_t)m * HID + col] = acc[r] + bv;
    }
}

// ---------------------------------------------------------------- launcher

extern "C" void kernel_launch(void* const* d_in, const int* in_sizes, int n_in,
                              void* d_out, int out_size, void* d_ws, size_t ws_size,
                              hipStream_t stream) {
    const float* x   = (const float*)d_in[0];   // [32,1024,1024]
    const float* Wi  = (const float*)d_in[1];   // [4,1024,1024]
    const float* Wh  = (const float*)d_in[2];   // [4,1024,1024]
    const float* b   = (const float*)d_in[3];   // [4,1024]
    const float* fcW = (const float*)d_in[4];   // [1024,1024]
    const float* fcb = (const float*)d_in[5];   // [1024]
    float* out = (float*)d_out;

    char* ws = (char*)d_ws;
    const size_t SEQ_BYTES = (size_t)BATCH * SEQT * HID * 2;    //  64 MB bf16
    const size_t XP_BYTES  = (size_t)BATCH * SEQT * HID * 4;    // 128 MB f32
    const size_t WP_BYTES  = (size_t)LAYERS * HID * HID * 2;    //   8 MB each
    unsigned short* seqb  = (unsigned short*)(ws);
    float*          xp    = (float*)(ws + SEQ_BYTES);
    unsigned short* WiP   = (unsigned short*)(ws + SEQ_BYTES + XP_BYTES);
    unsigned short* WhP   = (unsigned short*)(ws + SEQ_BYTES + XP_BYTES + WP_BYTES);
    unsigned short* fcWP  = (unsigned short*)(ws + SEQ_BYTES + XP_BYTES + 2 * WP_BYTES);
    unsigned short* hprev = (unsigned short*)(ws + SEQ_BYTES + XP_BYTES + 2 * WP_BYTES
                                                 + (size_t)HID * HID * 2);
    unsigned* bar = (unsigned*)((char*)hprev + (size_t)BATCH * HID * 2);

    const int nelem = BATCH * SEQT * HID;
    cvt_f32_bf16_kernel<<<4096, 256, 0, stream>>>(x, seqb, nelem);
    repack_b_kernel<<<4096, 256, 0, stream>>>(Wi, WiP, LAYERS);
    repack_b_kernel<<<4096, 256, 0, stream>>>(Wh, WhP, LAYERS);
    repack_b_kernel<<<2048, 256, 0, stream>>>(fcW, fcWP, 1);

    for (int l = 0; l < LAYERS; ++l) {
        proj_kernel<<<dim3(2048, 2), 256, 0, stream>>>(
            seqb, WiP + (size_t)l * HID * HID, b + (size_t)l * HID, xp);
        init_barrier_kernel<<<1, 32, 0, stream>>>(bar);
        scan_kernel<<<16, 256, 0, stream>>>(
            xp, WhP + (size_t)l * HID * HID, seqb, hprev, bar);
    }
    fc_kernel<<<16, 256, 0, stream>>>(seqb, fcWP, fcb, out);
    (void)in_sizes; (void)n_in; (void)out_size; (void)ws_size;
}
